// TransformerEncoder_13546326852244
// MI455X (gfx1250) — matrix operand order repaired and checked by reading, not compiled, not run on hardware
//
#include <hip/hip_runtime.h>
#include <hip/hip_bf16.h>
#include <math.h>

// ---------------- model dimensions ----------------
#define BB      8
#define NBINS   512
#define BINW    10
#define DMODEL  1024
#define HEADS   8
#define LAYERS  4
#define FF      1024
#define NEXP    30
#define TOPK    4
#define SEQ     513            // NB + 1
#define DHEAD   128
#define CAPACITY 1152
#define TT      (BB*SEQ)       // 4104 tokens
#define NB4     (BB*NBINS)     // 4096 bins
#define ATT_LD  520            // padded so bf16 rows stay 16B aligned

typedef unsigned short u16;
typedef unsigned int   u32;
typedef __attribute__((ext_vector_type(16))) __bf16 v16bf;
typedef __attribute__((ext_vector_type(8)))  float  v8f;

#define CDIV(a,b) (((a)+(b)-1)/(b))

__device__ __forceinline__ u16 f2bf(float f) {
    u32 u = __float_as_uint(f);
    u32 r = u + 0x7FFFu + ((u >> 16) & 1u);   // round-to-nearest-even
    return (u16)(r >> 16);
}
__device__ __forceinline__ float gelu_f(float x) {
    return 0.5f * x * (1.0f + erff(x * 0.70710678118654752f));
}
// async global->LDS copy (tracked on ASYNCcnt); offset applies to BOTH sides
__device__ __forceinline__ void async_lds_b128(u32 lds, const void* gsrc) {
    asm volatile("global_load_async_to_lds_b128 %0, %1, off"
                 :: "v"(lds), "v"(gsrc) : "memory");
}
__device__ __forceinline__ void async_lds_b128_o16(u32 lds, const void* gsrc) {
    asm volatile("global_load_async_to_lds_b128 %0, %1, off offset:16"
                 :: "v"(lds), "v"(gsrc) : "memory");
}
__device__ __forceinline__ void async_lds_b128_o32(u32 lds, const void* gsrc) {
    asm volatile("global_load_async_to_lds_b128 %0, %1, off offset:32"
                 :: "v"(lds), "v"(gsrc) : "memory");
}
__device__ __forceinline__ void async_lds_b128_o48(u32 lds, const void* gsrc) {
    asm volatile("global_load_async_to_lds_b128 %0, %1, off offset:48"
                 :: "v"(lds), "v"(gsrc) : "memory");
}
__device__ __forceinline__ void wait_async() {
    asm volatile("s_wait_asynccnt 0x0" ::: "memory");
}

// ---------------- generic bf16 WMMA GEMM (B always stored [N][K]) ----------
// C[z][M,N] = alpha * A[z][M,K] @ B[z][N,K]^T (+bias)(+gelu)(+residual)
// 128 threads = 4 waves; block tile 64(M) x 128(N); each wave 32x64
// -> 2 A-frags x 4 B-frags = 8 independent WMMAs per K-step per wave.
// LDS double-buffered; interior tiles staged with GLOBAL_LOAD_ASYNC_TO_LDS.
enum { GF_BIAS=1, GF_GELU=2, GF_RES=4, GF_F32=8, GF_BF16=16, GF_VT=32 };

union FragU { v16bf v; u32 u[8]; };

__global__ __launch_bounds__(128)
void gemm_bf16_wmma(const u16* __restrict__ A, long lda, long aOff1, long aOff2,
                    const u16* __restrict__ Bm, long ldb, long bOff1, long bOff2,
                    float* __restrict__ C, u16* __restrict__ Cb,
                    long ldc, long cOff1, long cOff2,
                    const float* __restrict__ bias, long biasOff1, long biasOff2,
                    const float* __restrict__ Res,
                    int M, int N, int Kd, int zdiv, float alpha, int flags)
{
    __shared__ u16 As[2][64][40];    // [buf][m][k], 80B pitch
    __shared__ u16 Bs[2][128][40];   // [buf][n][k]
    const int  t  = threadIdx.x;
    const long z  = blockIdx.z;
    const long z1 = z / zdiv, z2 = z % zdiv;
    const u16* Ab = A  + z1*aOff1 + z2*aOff2;
    const u16* Bb = Bm + z1*bOff1 + z2*bOff2;
    const long cbase = z1*cOff1 + z2*cOff2;
    const float* biasb = bias + z1*biasOff1 + z2*biasOff2;
    const int m0 = blockIdx.y * 64;
    const int n0 = blockIdx.x * 128;
    const int lane = t & 31, wave = t >> 5;
    const int mi = (wave >> 1) * 32, ni = (wave & 1) * 64;
    const int r = lane & 15, hi = lane >> 4;
    const int srow = t >> 1, skk = (t & 1) * 16;   // A staging: 16 halves/thread

    const bool interior = (m0 + 64 <= M) && (n0 + 128 <= N);

    // ---- staging helpers ----
    auto stageA_fast = [&](int buf, int k0) {   // 2x async b128 per thread
        const u16* src = Ab + (long)(m0 + srow) * lda + k0 + skk;
        u32 lds = (u32)(size_t)&As[buf][srow][skk];
        async_lds_b128(lds, src);
        async_lds_b128_o16(lds, src);
    };
    auto stageB_fast = [&](int buf, int k0) {   // 4x async b128 per thread
        const u16* src = Bb + (long)(n0 + t) * ldb + k0;
        u32 lds = (u32)(size_t)&Bs[buf][t][0];
        async_lds_b128(lds, src);
        async_lds_b128_o16(lds, src);
        async_lds_b128_o32(lds, src);
        async_lds_b128_o48(lds, src);
    };
    auto stageA_edge = [&](int buf, int k0) {
        int gm = m0 + srow;
        bool rowok = gm < M;
        const u16* src = Ab + (long)gm * lda + k0 + skk;
        for (int j = 0; j < 16; ++j) {
            int gk = k0 + skk + j;
            As[buf][srow][skk + j] = (rowok && gk < Kd) ? src[j] : (u16)0;
        }
    };
    auto stageB_edge = [&](int buf, int k0) {
        int gn = n0 + t;
        bool rowok = gn < N;
        const u16* src = Bb + (long)gn * ldb + k0;
        for (int j = 0; j < 32; ++j) {
            int gk = k0 + j;
            Bs[buf][t][j] = (rowok && gk < Kd) ? src[j] : (u16)0;
        }
    };

    const v8f vzero = {0.f,0.f,0.f,0.f,0.f,0.f,0.f,0.f};
    v8f acc[2][4];
#pragma unroll
    for (int i = 0; i < 2; ++i)
#pragma unroll
        for (int j = 0; j < 4; ++j) acc[i][j] = vzero;

    // stage first tile
    if (interior && 32 <= Kd) { stageA_fast(0, 0); stageB_fast(0, 0); }
    else                      { stageA_edge(0, 0); stageB_edge(0, 0); }

    int pp = 0;
    for (int k0 = 0; k0 < Kd; k0 += 32) {
        wait_async();          // own async copies done -> then barrier makes all visible
        __syncthreads();
        const int kn = k0 + 32;
        if (kn < Kd) {   // stage next tile into other buffer (overlaps WMMA)
            if (interior && kn + 32 <= Kd) { stageA_fast(1 - pp, kn); stageB_fast(1 - pp, kn); }
            else                           { stageA_edge(1 - pp, kn); stageB_edge(1 - pp, kn); }
        }

        // ---- fragments per ISA 7.12.2 (wave32)
        // A 16x32: lane row = lane&15; uint j -> K = (j<4? 2j : 16+2(j-4)) + 8*hi
        // B 32x16: lane col = lane&15; uint j -> K = (j<4? 2j : 16+2(j-4)) + 8*hi
        FragU fa[2], fb[4];
#pragma unroll
        for (int j = 0; j < 8; ++j) {
            int kA = ((j < 4) ? (j * 2) : (16 + (j - 4) * 2)) + hi * 8;
            int kB = ((j < 4) ? (j * 2) : (16 + (j - 4) * 2)) + hi * 8;
            fa[0].u[j] = *(const u32*)&As[pp][mi + r][kA];
            fa[1].u[j] = *(const u32*)&As[pp][mi + 16 + r][kA];
            fb[0].u[j] = *(const u32*)&Bs[pp][ni + r][kB];
            fb[1].u[j] = *(const u32*)&Bs[pp][ni + 16 + r][kB];
            fb[2].u[j] = *(const u32*)&Bs[pp][ni + 32 + r][kB];
            fb[3].u[j] = *(const u32*)&Bs[pp][ni + 48 + r][kB];
        }
#pragma unroll
        for (int sm = 0; sm < 2; ++sm)
#pragma unroll
            for (int sn = 0; sn < 4; ++sn)
                acc[sm][sn] = __builtin_amdgcn_wmma_f32_16x16x32_bf16(
                    false, fa[sm].v, false, fb[sn].v, (short)0, acc[sm][sn], false, false);
        pp ^= 1;
    }

    // ---- epilogue: D vgpr v -> row = v + 8*hi, col = lane&15
#pragma unroll
    for (int sm = 0; sm < 2; ++sm) {
#pragma unroll
        for (int sn = 0; sn < 4; ++sn) {
            int col = n0 + ni + sn * 16 + r;
            if (col >= N) continue;
#pragma unroll
            for (int v = 0; v < 8; ++v) {
                int row = m0 + mi + sm * 16 + hi * 8 + v;
                if (row >= M) continue;
                float x = acc[sm][sn][v] * alpha;
                if (flags & GF_BIAS) x += biasb[col];
                if (flags & GF_GELU) x = gelu_f(x);
                long idx = cbase + (long)row * ldc + col;
                if (flags & GF_RES)  x += Res[idx];
                if (flags & GF_F32)  C[idx]  = x;
                if (flags & GF_BF16) Cb[idx] = f2bf(x);
                if (flags & GF_VT) {   // V-projection: write [b][h][dh][seq] bf16
                    int bb = row / SEQ, s = row % SEQ;
                    int hh = col / DHEAD, dh = col % DHEAD;
                    Cb[(((long)bb*HEADS + hh)*DHEAD + dh)*ATT_LD + s] = f2bf(x);
                }
            }
        }
    }
}

// ---------------- transpose + convert: [batch][K][N] f32 -> [batch][N][K] bf16
__global__ __launch_bounds__(256)
void transpose_cvt_kernel(const float* __restrict__ in, u16* __restrict__ out,
                          int K, int N)
{
    __shared__ float tile[32][33];
    const int k0 = blockIdx.y * 32, n0 = blockIdx.x * 32;
    const long bo = (long)blockIdx.z * K * N;
    const int tx = threadIdx.x & 31, ty = threadIdx.x >> 5;   // 32 x 8
    for (int i = ty; i < 32; i += 8) {
        int k = k0 + i, n = n0 + tx;
        tile[i][tx] = (k < K && n < N) ? in[bo + (long)k * N + n] : 0.f;
    }
    __syncthreads();
    for (int i = ty; i < 32; i += 8) {
        int n = n0 + i, k = k0 + tx;
        if (n < N && k < K) out[bo + (long)n * K + k] = f2bf(tile[tx][i]);
    }
}

// ---------------- conv stem (4 bins per 128-thread block) ----------------
__global__ __launch_bounds__(128)
void stem_kernel(const float* __restrict__ x,
                 const float* __restrict__ w1, const float* __restrict__ w2,
                 const float* __restrict__ w3,
                 const float* __restrict__ g1, const float* __restrict__ b1,
                 const float* __restrict__ g2, const float* __restrict__ b2,
                 const float* __restrict__ g3, const float* __restrict__ b3,
                 u16* __restrict__ stem, int* __restrict__ msk)
{
    __shared__ float sx[4][BINW];
    __shared__ float s1[4][16][BINW];
    __shared__ float s2[4][32][BINW];
    const int sub = threadIdx.x >> 5, lane = threadIdx.x & 31;
    const int bin = blockIdx.x * 4 + sub;
    const float bns = 0.9999950000374997f;   // 1/sqrt(1+1e-5)

    if (lane < BINW) sx[sub][lane] = x[(long)bin * BINW + lane];
    __syncthreads();
    if (lane == 0) {
        int all = 1;
        for (int i = 0; i < BINW; ++i) all &= (sx[sub][i] == -100.0f);
        msk[bin] = all;
    }
    if (lane < 16) {   // conv1: 1->16 ch, k=3, dil=1
        float w0 = w1[lane*3+0], wA = w1[lane*3+1], wB = w1[lane*3+2];
        float sc = g1[lane]*bns, bi = b1[lane];
        for (int p = 0; p < BINW; ++p) {
            float a = (p-1 >= 0)   ? sx[sub][p-1] : 0.f;
            float b = sx[sub][p];
            float c = (p+1 < BINW) ? sx[sub][p+1] : 0.f;
            s1[sub][lane][p] = gelu_f((w0*a + wA*b + wB*c)*sc + bi);
        }
    }
    __syncthreads();
    {   // conv2: 16->32 ch, dil=2
        int c = lane;
        float sc = g2[c]*bns, bi = b2[c];
        for (int p = 0; p < BINW; ++p) {
            float y = 0.f;
            for (int ic = 0; ic < 16; ++ic) {
                const float* wp = w2 + ((long)c*16 + ic)*3;
                float a = (p-2 >= 0)   ? s1[sub][ic][p-2] : 0.f;
                float b = s1[sub][ic][p];
                float d = (p+2 < BINW) ? s1[sub][ic][p+2] : 0.f;
                y += wp[0]*a + wp[1]*b + wp[2]*d;
            }
            s2[sub][c][p] = gelu_f(y*sc + bi);
        }
    }
    __syncthreads();
    for (int cc = 0; cc < 2; ++cc) {   // conv3: 32->64 ch, dil=4, + mean pool
        int c = lane + cc*32;
        float sc = g3[c]*bns, bi = b3[c];
        float mean = 0.f;
        for (int p = 0; p < BINW; ++p) {
            float y = 0.f;
            for (int ic = 0; ic < 32; ++ic) {
                const float* wp = w3 + ((long)c*32 + ic)*3;
                float a = (p-4 >= 0)   ? s2[sub][ic][p-4] : 0.f;
                float b = s2[sub][ic][p];
                float d = (p+4 < BINW) ? s2[sub][ic][p+4] : 0.f;
                y += wp[0]*a + wp[1]*b + wp[2]*d;
            }
            mean += gelu_f(y*sc + bi);
        }
        stem[(long)bin*64 + c] = f2bf(mean * 0.1f);
    }
}

// ---------------- embedding assembly (+positional encoding) ----------------
__global__ void assemble_kernel(const float* __restrict__ projC, const int* __restrict__ msk,
                                const float* __restrict__ cls, const float* __restrict__ mtok,
                                float* __restrict__ h)
{
    long i = blockIdx.x*256L + threadIdx.x;
    if (i >= (long)BB*SEQ*DMODEL) return;
    int d = (int)(i % DMODEL);
    long bs = i / DMODEL;
    int s = (int)(bs % SEQ); int b = (int)(bs / SEQ);
    float v;
    if (s == 0) v = cls[d];
    else {
        long bin = (long)b*NBINS + (s-1);
        v = msk[bin] ? mtok[d] : projC[bin*DMODEL + d];
    }
    int   i2  = d & ~1;
    float div = __expf((float)i2 * (-9.210340371976184f / (float)DMODEL));
    float ang = (float)s * div;
    v += (d & 1) ? __cosf(ang) : __sinf(ang);
    h[i] = v;
}

// ---------------- LayerNorm (one row per block, bf16 out) ----------------
__global__ __launch_bounds__(256)
void ln_kernel(const float* __restrict__ h, const float* __restrict__ g,
               const float* __restrict__ b, u16* __restrict__ out)
{
    __shared__ float red[256];
    const int row = blockIdx.x;
    const float* hr = h + (long)row * DMODEL;
    int t = threadIdx.x;
    float s = 0.f;
    for (int d = t; d < DMODEL; d += 256) s += hr[d];
    red[t] = s; __syncthreads();
    for (int o = 128; o > 0; o >>= 1) { if (t < o) red[t] += red[t+o]; __syncthreads(); }
    float mean = red[0] * (1.0f/DMODEL);
    __syncthreads();
    float v = 0.f;
    for (int d = t; d < DMODEL; d += 256) { float q = hr[d]-mean; v += q*q; }
    red[t] = v; __syncthreads();
    for (int o = 128; o > 0; o >>= 1) { if (t < o) red[t] += red[t+o]; __syncthreads(); }
    float rs = rsqrtf(red[0]*(1.0f/DMODEL) + 1e-5f);
    for (int d = t; d < DMODEL; d += 256)
        out[(long)row*DMODEL + d] = f2bf((hr[d]-mean)*rs*g[d] + b[d]);
}

// ---------------- attention softmax (one score row per block) ----------------
__global__ __launch_bounds__(256)
void softmax_kernel(const float* __restrict__ scores, u16* __restrict__ att)
{
    __shared__ float red[256];
    const long row = blockIdx.x;                  // BH*SEQ rows
    const float* sr = scores + row * SEQ;
    u16* ar = att + row * ATT_LD;
    int t = threadIdx.x;
    float mx = -1e30f;
    for (int i = t; i < SEQ; i += 256) mx = fmaxf(mx, sr[i]);
    red[t] = mx; __syncthreads();
    for (int o=128;o>0;o>>=1){ if(t<o) red[t]=fmaxf(red[t],red[t+o]); __syncthreads(); }
    mx = red[0]; __syncthreads();
    float s = 0.f;
    for (int i = t; i < SEQ; i += 256) s += __expf(sr[i]-mx);
    red[t] = s; __syncthreads();
    for (int o=128;o>0;o>>=1){ if(t<o) red[t]+=red[t+o]; __syncthreads(); }
    float inv = 1.f/red[0];
    for (int i = t; i < ATT_LD; i += 256)
        ar[i] = (i < SEQ) ? f2bf(__expf(sr[i]-mx)*inv) : (u16)0;
}

// ---------------- MoE routing: softmax + top4 + capacity slots ----------------
__global__ void route_kernel(const float* __restrict__ logits, int* __restrict__ cnt,
                             int* __restrict__ tok_e, int* __restrict__ tok_pos,
                             float* __restrict__ tok_gate)
{
    int tkn = blockIdx.x*256 + threadIdx.x;
    if (tkn >= TT) return;
    float l[NEXP];
    float mx = -1e30f;
    for (int e = 0; e < NEXP; ++e) { l[e] = logits[(long)tkn*NEXP + e]; mx = fmaxf(mx, l[e]); }
    float s = 0.f;
    for (int e = 0; e < NEXP; ++e) { l[e] = __expf(l[e]-mx); s += l[e]; }
    float inv = 1.f/s;
    int idx[TOPK]; float g[TOPK]; float gs = 0.f;
    for (int kk = 0; kk < TOPK; ++kk) {
        int bi = 0; float bv = -1.f;
        for (int e = 0; e < NEXP; ++e) if (l[e] > bv) { bv = l[e]; bi = e; }
        idx[kk] = bi; g[kk] = bv*inv; gs += g[kk]; l[bi] = -1.f;
    }
    float ginv = 1.f/gs;
    for (int kk = 0; kk < TOPK; ++kk) {
        int e = idx[kk];
        int pos = atomicAdd(&cnt[e], 1);
        int o = tkn*TOPK + kk;
        tok_e[o]    = e;
        tok_pos[o]  = (pos < CAPACITY) ? pos : -1;
        tok_gate[o] = g[kk]*ginv;
    }
}

// ---------------- MoE dispatch: scatter bf16 token rows into expert buffers ----
__global__ __launch_bounds__(128)
void dispatch_kernel(const u16* __restrict__ xt, const int* __restrict__ tok_e,
                     const int* __restrict__ tok_pos, u16* __restrict__ buf)
{
    int o = blockIdx.x;             // T*K slots
    int pos = tok_pos[o];
    if (pos < 0) return;
    int e = tok_e[o];
    int tkn = o / TOPK;
    const u32* src = (const u32*)(xt + (long)tkn*DMODEL);
    u32* dst = (u32*)(buf + ((long)e*CAPACITY + pos)*DMODEL);
    for (int i = threadIdx.x; i < DMODEL/2; i += 128) dst[i] = src[i];
}

// ---------------- MoE combine: h += sum_k gate*expert_out ----------------
__global__ void combine_kernel(const float* __restrict__ expout,
                               const int* __restrict__ tok_e, const int* __restrict__ tok_pos,
                               const float* __restrict__ tok_gate, float* __restrict__ h)
{
    long i = blockIdx.x*256L + threadIdx.x;
    if (i >= (long)TT*DMODEL) return;
    int d = (int)(i % DMODEL);
    int tkn = (int)(i / DMODEL);
    float acc = 0.f;
    for (int kk = 0; kk < TOPK; ++kk) {
        int o = tkn*TOPK + kk;
        int pos = tok_pos[o];
        if (pos >= 0)
            acc += tok_gate[o] * expout[((long)tok_e[o]*CAPACITY + pos)*DMODEL + d];
    }
    h[i] += acc;
}

// ---------------- helpers ----------------
__global__ void zero_u32_kernel(u32* __restrict__ p, long n) {
    long i = blockIdx.x*256L + threadIdx.x;
    if (i < n) p[i] = 0u;
}
__global__ void out_kernel(const float* __restrict__ h, float* __restrict__ out) {
    int i = blockIdx.x*256 + threadIdx.x;
    if (i >= BB*DMODEL) return;
    int b = i / DMODEL, d = i % DMODEL;
    out[i] = h[(long)b*SEQ*DMODEL + d];
}

// ---------------- host-side launch helpers ----------------
static void gemm(hipStream_t st,
                 const u16* A, long lda, long a1, long a2,
                 const u16* Bm, long ldb, long b1, long b2,
                 float* C, u16* Cb, long ldc, long c1, long c2,
                 const float* bias, long bi1, long bi2,
                 const float* Res, int M, int N, int Kd,
                 int nz, int zdiv, float alpha, int flags)
{
    dim3 g(CDIV(N,128), CDIV(M,64), nz);
    gemm_bf16_wmma<<<g, 128, 0, st>>>(A, lda, a1, a2, Bm, ldb, b1, b2,
                                      C, Cb, ldc, c1, c2, bias, bi1, bi2, Res,
                                      M, N, Kd, zdiv, alpha, flags);
}
static void tcvt(hipStream_t st, const float* in, u16* out, int K, int N, int batch) {
    dim3 g(CDIV(N,32), CDIV(K,32), batch);
    transpose_cvt_kernel<<<g, 256, 0, st>>>(in, out, K, N);
}

extern "C" void kernel_launch(void* const* d_in, const int* in_sizes, int n_in,
                              void* d_out, int out_size, void* d_ws, size_t ws_size,
                              hipStream_t stream)
{
    (void)in_sizes; (void)n_in; (void)out_size; (void)ws_size;
    // ---- inputs (setup_inputs order) ----
    const float* x      = (const float*)d_in[0];
    const float* cw1    = (const float*)d_in[1];
    const float* cw2    = (const float*)d_in[2];
    const float* cw3    = (const float*)d_in[3];
    const float* bn1g   = (const float*)d_in[4];
    const float* bn1b   = (const float*)d_in[5];
    const float* bn2g   = (const float*)d_in[6];
    const float* bn2b   = (const float*)d_in[7];
    const float* bn3g   = (const float*)d_in[8];
    const float* bn3b   = (const float*)d_in[9];
    const float* projw  = (const float*)d_in[10];
    const float* projb  = (const float*)d_in[11];
    const float* cls    = (const float*)d_in[12];
    const float* mtok   = (const float*)d_in[13];
    const float* ln1g   = (const float*)d_in[14];
    const float* ln1b   = (const float*)d_in[15];
    const float* ln2g   = (const float*)d_in[16];
    const float* ln2b   = (const float*)d_in[17];
    const float* wq     = (const float*)d_in[18];
    const float* wk     = (const float*)d_in[19];
    const float* wv     = (const float*)d_in[20];
    const float* wo     = (const float*)d_in[21];
    const float* bq     = (const float*)d_in[22];
    const float* bk     = (const float*)d_in[23];
    const float* bv     = (const float*)d_in[24];
    const float* bo     = (const float*)d_in[25];
    const float* wg     = (const float*)d_in[26];
    const float* bg     = (const float*)d_in[27];
    const float* we1    = (const float*)d_in[28];
    const float* be1    = (const float*)d_in[29];
    const float* we2    = (const float*)d_in[30];
    const float* be2    = (const float*)d_in[31];
    float* out = (float*)d_out;

    // ---- workspace carve ----
    size_t offs = 0;
    char* base = (char*)d_ws;
    auto alloc = [&](size_t bytes) -> void* {
        void* p = base + offs;
        offs = (offs + bytes + 255) & ~(size_t)255;
        return p;
    };
    const long LDD = (long)LAYERS*DMODEL*DMODEL;
    u16* wqT     = (u16*)alloc(LDD*2);                       // [L][D][D] -> [L][N=D][K=D]
    u16* wkT     = (u16*)alloc(LDD*2);
    u16* wvT     = (u16*)alloc(LDD*2);
    u16* woT     = (u16*)alloc(LDD*2);
    u16* wgT     = (u16*)alloc((long)LAYERS*NEXP*DMODEL*2);  // [L][E][D]
    u16* projwT  = (u16*)alloc((long)DMODEL*64*2);           // [D][64]
    u16* we1T    = (u16*)alloc((long)NEXP*FF*DMODEL*2);      // per-layer [E][F][D]
    u16* we2T    = (u16*)alloc((long)NEXP*DMODEL*FF*2);      // per-layer [E][D][F]
    u16* stem_bf = (u16*)alloc((long)NB4*64*2);
    int* msk     = (int*)alloc((long)NB4*4);
    float* projC = (float*)alloc((long)NB4*DMODEL*4);
    float* h     = (float*)alloc((long)TT*DMODEL*4);
    u16* aBuf    = (u16*)alloc((long)TT*DMODEL*2);
    u16* a2Buf   = (u16*)alloc((long)TT*DMODEL*2);
    u16* qB      = (u16*)alloc((long)TT*DMODEL*2);
    u16* kB      = (u16*)alloc((long)TT*DMODEL*2);
    u16* vT      = (u16*)alloc((long)BB*HEADS*DHEAD*ATT_LD*2);  // [b][h][dh][seq]
    float* scores= (float*)alloc((long)BB*HEADS*SEQ*SEQ*4);
    u16* att     = (u16*)alloc((long)BB*HEADS*SEQ*ATT_LD*2);
    u16* ctx     = (u16*)alloc((long)TT*DMODEL*2);
    float* logits= (float*)alloc((long)TT*NEXP*4);
    int* cnt     = (int*)alloc((long)NEXP*4);
    int* tok_e   = (int*)alloc((long)TT*TOPK*4);
    int* tok_pos = (int*)alloc((long)TT*TOPK*4);
    float* tok_g = (float*)alloc((long)TT*TOPK*4);
    u16* moebuf  = (u16*)alloc((long)NEXP*CAPACITY*DMODEL*2);
    u16* moehid  = (u16*)alloc((long)NEXP*CAPACITY*FF*2);
    float* expout= (float*)alloc((long)NEXP*CAPACITY*DMODEL*4);

    // ---- transpose+convert all weights to [N][K] bf16 ----
    tcvt(stream, wq, wqT, DMODEL, DMODEL, LAYERS);
    tcvt(stream, wk, wkT, DMODEL, DMODEL, LAYERS);
    tcvt(stream, wv, wvT, DMODEL, DMODEL, LAYERS);
    tcvt(stream, wo, woT, DMODEL, DMODEL, LAYERS);
    tcvt(stream, wg, wgT, DMODEL, NEXP, LAYERS);
    tcvt(stream, projw, projwT, 64, DMODEL, 1);

    // ---- stem + projection + embedding ----
    stem_kernel<<<NB4/4, 128, 0, stream>>>(x, cw1, cw2, cw3, bn1g, bn1b,
                                           bn2g, bn2b, bn3g, bn3b, stem_bf, msk);
    gemm(stream, stem_bf, 64, 0,0, projwT, 64, 0,0,
         projC, nullptr, DMODEL, 0,0, projb, 0,0, nullptr,
         NB4, DMODEL, 64, 1, 1, 1.0f, GF_BIAS|GF_F32);
    assemble_kernel<<<(int)CDIV((long)TT*DMODEL,256), 256, 0, stream>>>(projC, msk, cls, mtok, h);

    const float att_scale = 0.08838834764831845f;   // 1/sqrt(128)
    for (int l = 0; l < LAYERS; ++l) {
        const u16* Wq = wqT + (long)l*DMODEL*DMODEL;
        const u16* Wk = wkT + (long)l*DMODEL*DMODEL;
        const u16* Wv = wvT + (long)l*DMODEL*DMODEL;
        const u16* Wo = woT + (long)l*DMODEL*DMODEL;

        // LN1 -> a (bf16)
        ln_kernel<<<TT, 256, 0, stream>>>(h, ln1g + (long)l*DMODEL, ln1b + (long)l*DMODEL, aBuf);

        // Q/K/V projections (V written transposed to [b][h][dh][seq])
        gemm(stream, aBuf, DMODEL, 0,0, Wq, DMODEL, 0,0, nullptr, qB, DMODEL, 0,0,
             bq + (long)l*DMODEL, 0,0, nullptr, TT, DMODEL, DMODEL, 1,1, 1.0f, GF_BIAS|GF_BF16);
        gemm(stream, aBuf, DMODEL, 0,0, Wk, DMODEL, 0,0, nullptr, kB, DMODEL, 0,0,
             bk + (long)l*DMODEL, 0,0, nullptr, TT, DMODEL, DMODEL, 1,1, 1.0f, GF_BIAS|GF_BF16);
        gemm(stream, aBuf, DMODEL, 0,0, Wv, DMODEL, 0,0, nullptr, vT, DMODEL, 0,0,
             bv + (long)l*DMODEL, 0,0, nullptr, TT, DMODEL, DMODEL, 1,1, 1.0f, GF_BIAS|GF_VT);

        // scores[b,h] = scale * Q[b,:,h,:] @ K[b,:,h,:]^T  (batched over b*h)
        gemm(stream, qB, DMODEL, (long)SEQ*DMODEL, DHEAD,
                     kB, DMODEL, (long)SEQ*DMODEL, DHEAD,
             scores, nullptr, SEQ, (long)HEADS*SEQ*SEQ, (long)SEQ*SEQ,
             nullptr, 0,0, nullptr, SEQ, SEQ, DHEAD, BB*HEADS, HEADS, att_scale, GF_F32);

        softmax_kernel<<<BB*HEADS*SEQ, 256, 0, stream>>>(scores, att);

        // ctx[b,:,h,:] = att[b,h] @ V[b,h]   (B = vT rows [dh][seq], row-major)
        gemm(stream, att, ATT_LD, (long)HEADS*SEQ*ATT_LD, (long)SEQ*ATT_LD,
                     vT, ATT_LD, (long)HEADS*DHEAD*ATT_LD, (long)DHEAD*ATT_LD,
             nullptr, ctx, DMODEL, (long)SEQ*DMODEL, DHEAD,
             nullptr, 0,0, nullptr, SEQ, DHEAD, SEQ, BB*HEADS, HEADS, 1.0f, GF_BF16);

        // h = h + ctx @ Wo + bo
        gemm(stream, ctx, DMODEL, 0,0, Wo, DMODEL, 0,0, h, nullptr, DMODEL, 0,0,
             bo + (long)l*DMODEL, 0,0, h, TT, DMODEL, DMODEL, 1,1, 1.0f, GF_BIAS|GF_RES|GF_F32);

        // LN2 -> a2 (bf16)
        ln_kernel<<<TT, 256, 0, stream>>>(h, ln2g + (long)l*DMODEL, ln2b + (long)l*DMODEL, a2Buf);

        // router logits
        gemm(stream, a2Buf, DMODEL, 0,0, wgT + (long)l*NEXP*DMODEL, DMODEL, 0,0,
             logits, nullptr, NEXP, 0,0, bg + (long)l*NEXP, 0,0, nullptr,
             TT, NEXP, DMODEL, 1,1, 1.0f, GF_BIAS|GF_F32);

        // routing + dispatch (buffers re-zeroed each call for determinism)
        zero_u32_kernel<<<1, 256, 0, stream>>>((u32*)cnt, NEXP);
        zero_u32_kernel<<<(int)CDIV((long)NEXP*CAPACITY*DMODEL/2,256), 256, 0, stream>>>(
            (u32*)moebuf, (long)NEXP*CAPACITY*DMODEL/2);
        route_kernel<<<CDIV(TT,256), 256, 0, stream>>>(logits, cnt, tok_e, tok_pos, tok_g);
        dispatch_kernel<<<TT*TOPK, 128, 0, stream>>>(a2Buf, tok_e, tok_pos, moebuf);

        // per-layer expert weights -> transposed bf16
        tcvt(stream, we1 + (long)l*NEXP*DMODEL*FF, we1T, DMODEL, FF, NEXP);   // [E][F][D]
        tcvt(stream, we2 + (long)l*NEXP*FF*DMODEL, we2T, FF, DMODEL, NEXP);   // [E][D][F]

        // expert FFN: hid = gelu(buf @ W1 + b1) ; out = hid @ W2 + b2 (batched over experts)
        gemm(stream, moebuf, DMODEL, (long)CAPACITY*DMODEL, 0,
                     we1T, DMODEL, (long)FF*DMODEL, 0,
             nullptr, moehid, FF, (long)CAPACITY*FF, 0,
             be1 + (long)l*NEXP*FF, FF, 0, nullptr,
             CAPACITY, FF, DMODEL, NEXP, 1, 1.0f, GF_BIAS|GF_GELU|GF_BF16);
        gemm(stream, moehid, FF, (long)CAPACITY*FF, 0,
                     we2T, FF, (long)DMODEL*FF, 0,
             expout, nullptr, DMODEL, (long)CAPACITY*DMODEL, 0,
             be2 + (long)l*NEXP*DMODEL, DMODEL, 0, nullptr,
             CAPACITY, DMODEL, FF, NEXP, 1, 1.0f, GF_BIAS|GF_F32);

        combine_kernel<<<(int)CDIV((long)TT*DMODEL,256), 256, 0, stream>>>(
            expout, tok_e, tok_pos, tok_g, h);
    }

    out_kernel<<<CDIV(BB*DMODEL,256), 256, 0, stream>>>(h, out);
}
